// TemporalGCN_87952340287522
// MI455X (gfx1250) — matrix operand order repaired and checked by reading, not compiled, not run on hardware
//
#include <hip/hip_runtime.h>
#include <hip/hip_bf16.h>

#define TT 12
#define NN 50000
#define FF 32
#define HH 64
#define EE 1600000

typedef __attribute__((ext_vector_type(16))) __bf16 v16bf;
typedef __attribute__((ext_vector_type(8)))  float  v8f;
typedef __attribute__((ext_vector_type(4)))  float  v4f;
typedef __attribute__((ext_vector_type(4)))  unsigned int v4u;

static __device__ inline v8f wmma_bf16(v16bf a, v16bf b, v8f c) {
  // D = A(16x32 bf16) x B(32x16 bf16) + C(16x16 f32)
  return __builtin_amdgcn_wmma_f32_16x16x32_bf16(false, a, false, b, (short)0, c,
                                                 false, false);
}

// ---- A fragment: 16x32 bf16 from f32 row-major [16, lda], fused ReLU --------
// Per ISA 7.12.2: lane m=lane&15; lanes16-31 shift K by +8; VGPR v holds K pairs
// {2v, 2v+1} (+8 for v>=4).  Per lane that is two contiguous 8-float chunks:
// [kbase+8*hi .. +7] and [kbase+8*hi+16 .. +23]  -> four float4 loads.
static __device__ inline v16bf load_A_frag_f32(const float* __restrict__ Arow,
                                               int lda, int kbase, int relu) {
  int lane = threadIdx.x & 31;
  int m    = lane & 15;
  int hi   = lane >> 4;
  const float* r = Arow + (size_t)m * lda + kbase + hi * 8;
  v4f c0 = *(const v4f*)(r);
  v4f c1 = *(const v4f*)(r + 4);
  v4f c2 = *(const v4f*)(r + 16);
  v4f c3 = *(const v4f*)(r + 20);
  float x[16];
#pragma unroll
  for (int i = 0; i < 4; ++i) {
    x[i] = c0[i]; x[4 + i] = c1[i]; x[8 + i] = c2[i]; x[12 + i] = c3[i];
  }
  v16bf a;
#pragma unroll
  for (int i = 0; i < 16; ++i) {
    float v = x[i];
    if (relu) v = fmaxf(v, 0.f);
    a[i] = (__bf16)v;
  }
  return a;
}

// ---- B fragment from pre-packed buffer: [frag][lane][16 bf16] ---------------
// 32 bytes per lane, contiguous -> two global_load_b128, no shuffles.
static __device__ inline v16bf load_B_packed(const __bf16* __restrict__ Bpk, int frag) {
  int lane = threadIdx.x & 31;
  const v4u* p = (const v4u*)(Bpk + ((size_t)frag * 32 + lane) * 16);
  union { v4u u[2]; v16bf b; } u;
  u.u[0] = p[0];
  u.u[1] = p[1];
  return u.b;
}

// Pack weights into fragment order.  Source is f32:
//   transposed==0 : W is [K, NT*16] row-major (element W[k*NT16 + n])
//   transposed==1 : W is [N, K]     row-major (element W[n*K + k])   (W_ih / W_hh)
__global__ void pack_B_k(const float* __restrict__ W, __bf16* __restrict__ out,
                         int NT, int K, int transposed) {
  int i = blockIdx.x * blockDim.x + threadIdx.x;
  int total = (K / 32) * NT * 32 * 16;
  if (i >= total) return;
  int elem  = i & 15;
  int lane  = (i >> 4) & 31;
  int frag  = i >> 9;
  int ntile = frag % NT;
  int kstep = frag / NT;
  int n  = ntile * 16 + (lane & 15);
  int hi = lane >> 4;
  int v  = elem >> 1;
  int k  = kstep * 32 + hi * 8 + (elem & 7) + (elem >> 3) * 16;
  float val = transposed ? W[(size_t)n * K + k] : W[(size_t)k * (NT * 16) + n];
  out[i] = (__bf16)val;
}

// ---- generic WMMA GEMM: C[M, NT*16] = (relu?)A[M, KSTEPS*32] * B + bias -----
template <int NT, int KSTEPS>
__global__ __launch_bounds__(256) void gemm_wmma_k(const float* __restrict__ A,
                                                   const __bf16* __restrict__ Bpk,
                                                   const float* __restrict__ bias,
                                                   float* __restrict__ C,
                                                   int Mtiles, int relu_in) {
  const int lda = KSTEPS * 32;
  const int ldc = NT * 16;
  int wave  = threadIdx.x >> 5;
  int mtile = blockIdx.x * 8 + wave;
  if (mtile >= Mtiles) return;
  int lane = threadIdx.x & 31;
  int n15  = lane & 15, hi = lane >> 4;
  const float* Arow = A + (size_t)mtile * 16 * lda;
  v8f acc[NT] = {};
#pragma unroll
  for (int ks = 0; ks < KSTEPS; ++ks) {
    v16bf a = load_A_frag_f32(Arow, lda, ks * 32, relu_in);
#pragma unroll
    for (int t = 0; t < NT; ++t) {
      v16bf b = load_B_packed(Bpk, ks * NT + t);
      acc[t] = wmma_bf16(a, b, acc[t]);
    }
  }
#pragma unroll
  for (int t = 0; t < NT; ++t) {
    int col  = t * 16 + n15;
    float bv = bias ? bias[col] : 0.f;
#pragma unroll
    for (int j = 0; j < 8; ++j) {
      int row = mtile * 16 + j + hi * 8;      // C layout: VGPR j -> M=j / j+8
      C[(size_t)row * ldc + col] = acc[t][j] + bv;
    }
  }
}

// ---- graph normalization ----------------------------------------------------
__global__ void deg_init_k(float* deg) {
  int i = blockIdx.x * blockDim.x + threadIdx.x;
  if (i < NN) deg[i] = 1.0f;                  // +1 self loop
}
__global__ void deg_accum_k(const int* __restrict__ dst, float* deg) {
  int e = blockIdx.x * blockDim.x + threadIdx.x;
  if (e < EE) atomicAdd(&deg[dst[e]], 1.0f);
}
__global__ void deg_fin_k(const float* __restrict__ deg, float* dinv, float* invdeg) {
  int i = blockIdx.x * blockDim.x + threadIdx.x;
  if (i < NN) {
    float d = deg[i];
    dinv[i]   = rsqrtf(d);
    invdeg[i] = 1.0f / d;
  }
}
__global__ void norm_k(const int* __restrict__ src, const int* __restrict__ dst,
                       const float* __restrict__ dinv, float* norm) {
  int e = blockIdx.x * blockDim.x + threadIdx.x;
  if (e < EE) norm[e] = dinv[src[e]] * dinv[dst[e]];
}

// ---- GCN aggregation --------------------------------------------------------
// out[t,n,h] = xw[t,n,h]*invdeg[n] + b[h]   (then edges atomically add)
__global__ void self_init_k(const float* __restrict__ xw,
                            const float* __restrict__ invdeg,
                            const float* __restrict__ b, float* __restrict__ out) {
  size_t idx = (size_t)blockIdx.x * blockDim.x + threadIdx.x;
  if (idx >= (size_t)TT * NN * HH) return;
  int node = (int)((idx >> 6) % NN);
  out[idx] = xw[idx] * invdeg[node] + b[idx & 63];
}
// one 64-thread group per edge covers all H channels, loops T
__global__ void edge_scatter_k(const float* __restrict__ xw,
                               const int* __restrict__ src,
                               const int* __restrict__ dst,
                               const float* __restrict__ norm,
                               float* __restrict__ out) {
  size_t idx = (size_t)blockIdx.x * blockDim.x + threadIdx.x;
  int e = (int)(idx >> 6);
  int h = (int)(idx & 63);
  if (e >= EE) return;
  int s = src[e], d = dst[e];
  float nm = norm[e];
  const float* xs = xw + (size_t)s * HH + h;
  float*       od = out + (size_t)d * HH + h;
#pragma unroll
  for (int t = 0; t < TT; ++t) {
    atomicAdd(od + (size_t)t * NN * HH, xs[(size_t)t * NN * HH] * nm);
  }
}

// ---- GRU --------------------------------------------------------------------
__global__ void h_init_k(float* h) {
  int i = blockIdx.x * blockDim.x + threadIdx.x;
  if (i < NN * HH) h[i] = 0.0f;
}

static __device__ inline float sigmoidf(float x) { return 1.0f / (1.0f + expf(-x)); }

// one wave: gh[16,192] = h[16,64] @ WhhT[64,192] (WMMA) then lane-local gates
__global__ __launch_bounds__(256) void gru_step_k(const float* __restrict__ gi_t,
                                                  const __bf16* __restrict__ WhhPk,
                                                  const float* __restrict__ b_hh,
                                                  float* __restrict__ h, int Mtiles) {
  int wave  = threadIdx.x >> 5;
  int mtile = blockIdx.x * 8 + wave;
  if (mtile >= Mtiles) return;
  int lane = threadIdx.x & 31;
  int n15  = lane & 15, hi = lane >> 4;
  const float* Arow = h + (size_t)mtile * 16 * HH;
  v8f acc[12] = {};
#pragma unroll
  for (int ks = 0; ks < 2; ++ks) {
    v16bf a = load_A_frag_f32(Arow, HH, ks * 32, 0);
#pragma unroll
    for (int t = 0; t < 12; ++t) {
      v16bf b = load_B_packed(WhhPk, ks * 12 + t);
      acc[t] = wmma_bf16(a, b, acc[t]);
    }
  }
#pragma unroll
  for (int c = 0; c < 4; ++c) {               // gate columns 0..63
    int col  = c * 16 + n15;
    float br = b_hh[col], bz = b_hh[col + 64], bn = b_hh[col + 128];
#pragma unroll
    for (int j = 0; j < 8; ++j) {
      int node = mtile * 16 + j + hi * 8;
      const float* g = gi_t + (size_t)node * 192;
      float hr = acc[c][j]     + br;
      float hz = acc[c + 4][j] + bz;
      float hn = acc[c + 8][j] + bn;
      float r  = sigmoidf(g[col] + hr);
      float z  = sigmoidf(g[col + 64] + hz);
      float nn = tanhf(g[col + 128] + r * hn);
      float hp = h[(size_t)node * HH + col];
      h[(size_t)node * HH + col] = (1.0f - z) * nn + z * hp;
    }
  }
}

// ---- classifier: H->32 relu ->1 (tiny, VALU) --------------------------------
__global__ void classifier_k(const float* __restrict__ h, const float* __restrict__ Wc1,
                             const float* __restrict__ bc1, const float* __restrict__ Wc2,
                             const float* __restrict__ bc2, float* __restrict__ out) {
  int n = blockIdx.x * blockDim.x + threadIdx.x;
  if (n >= NN) return;
  const float* hr = h + (size_t)n * HH;
  float o = 0.0f;
  for (int j = 0; j < 32; ++j) {
    float s = bc1[j];
#pragma unroll
    for (int k = 0; k < HH; ++k) s += hr[k] * Wc1[k * 32 + j];
    s = fmaxf(s, 0.0f);
    o += s * Wc2[j];
  }
  out[n] = o + bc2[0];
}

extern "C" void kernel_launch(void* const* d_in, const int* in_sizes, int n_in,
                              void* d_out, int out_size, void* d_ws, size_t ws_size,
                              hipStream_t stream) {
  const float* x_seq = (const float*)d_in[0];
  const int*   eidx  = (const int*)d_in[1];
  const int*   src   = eidx;
  const int*   dst   = eidx + EE;
  const float* W1    = (const float*)d_in[2];
  const float* b1    = (const float*)d_in[3];
  const float* W2    = (const float*)d_in[4];
  const float* b2    = (const float*)d_in[5];
  const float* W_ih  = (const float*)d_in[6];
  const float* W_hh  = (const float*)d_in[7];
  const float* b_ih  = (const float*)d_in[8];
  const float* b_hh  = (const float*)d_in[9];
  const float* Wc1   = (const float*)d_in[10];
  const float* bc1   = (const float*)d_in[11];
  const float* Wc2   = (const float*)d_in[12];
  const float* bc2   = (const float*)d_in[13];
  float* out = (float*)d_out;

  char*  ws  = (char*)d_ws;
  size_t off = 0;
  auto alloc = [&](size_t bytes) -> void* {
    void* p = ws + off;
    off += (bytes + 255) & ~(size_t)255;
    return p;
  };

  const size_t TN = (size_t)TT * NN;          // 600000 rows
  float*  gi     = (float*)alloc(TN * 192 * sizeof(float));  // also aliases xw
  float*  outbuf = (float*)alloc(TN * HH * sizeof(float));
  float*  h      = (float*)alloc((size_t)NN * HH * sizeof(float));
  float*  deg    = (float*)alloc(NN * sizeof(float));
  float*  dinv   = (float*)alloc(NN * sizeof(float));
  float*  invdeg = (float*)alloc(NN * sizeof(float));
  float*  norm   = (float*)alloc(EE * sizeof(float));
  __bf16* W1pk   = (__bf16*)alloc((size_t)1 * 4 * 32 * 16 * sizeof(__bf16));   // 2048
  __bf16* W2pk   = (__bf16*)alloc((size_t)2 * 4 * 32 * 16 * sizeof(__bf16));   // 4096
  __bf16* WihPk  = (__bf16*)alloc((size_t)2 * 12 * 32 * 16 * sizeof(__bf16));  // 12288
  __bf16* WhhPk  = (__bf16*)alloc((size_t)2 * 12 * 32 * 16 * sizeof(__bf16));  // 12288
  float*  xw     = gi;                        // xw [TN,64] lives inside gi region

  const int Mtiles   = (int)(TN / 16);        // 37500
  const int gemmBlks = (Mtiles + 7) / 8;      // 4688
  const int selfBlks = (int)((TN * HH + 255) / 256);
  const int edgeBlks = (int)(((size_t)EE * HH + 255) / 256);

  // weight pre-packing into WMMA fragment order (frag-major, 32B/lane)
  pack_B_k<<<(2048 + 255) / 256, 256, 0, stream>>>(W1, W1pk, 4, 32, 0);
  pack_B_k<<<(4096 + 255) / 256, 256, 0, stream>>>(W2, W2pk, 4, 64, 0);
  pack_B_k<<<(12288 + 255) / 256, 256, 0, stream>>>(W_ih, WihPk, 12, 64, 1);
  pack_B_k<<<(12288 + 255) / 256, 256, 0, stream>>>(W_hh, WhhPk, 12, 64, 1);

  // gcn normalization
  deg_init_k<<<(NN + 255) / 256, 256, 0, stream>>>(deg);
  deg_accum_k<<<(EE + 255) / 256, 256, 0, stream>>>(dst, deg);
  deg_fin_k<<<(NN + 255) / 256, 256, 0, stream>>>(deg, dinv, invdeg);
  norm_k<<<(EE + 255) / 256, 256, 0, stream>>>(src, dst, dinv, norm);

  // ---- GCN layer 1: xw = x_seq @ W1 ; out = agg(xw) + xw*invdeg + b1
  gemm_wmma_k<4, 1><<<gemmBlks, 256, 0, stream>>>(x_seq, W1pk, nullptr, xw, Mtiles, 0);
  self_init_k<<<selfBlks, 256, 0, stream>>>(xw, invdeg, b1, outbuf);
  edge_scatter_k<<<edgeBlks, 256, 0, stream>>>(xw, src, dst, norm, outbuf);

  // ---- GCN layer 2 (ReLU of layer-1 output fused into A loader)
  gemm_wmma_k<4, 2><<<gemmBlks, 256, 0, stream>>>(outbuf, W2pk, nullptr, xw, Mtiles, 1);
  self_init_k<<<selfBlks, 256, 0, stream>>>(xw, invdeg, b2, outbuf);
  edge_scatter_k<<<edgeBlks, 256, 0, stream>>>(xw, src, dst, norm, outbuf);

  // ---- GRU input gates: gi = relu(outbuf) @ W_ih^T + b_ih   [TN, 192]
  gemm_wmma_k<12, 2><<<gemmBlks, 256, 0, stream>>>(outbuf, WihPk, b_ih, gi, Mtiles, 1);

  // ---- GRU recurrence
  h_init_k<<<(NN * HH + 255) / 256, 256, 0, stream>>>(h);
  const int gruTiles = NN / 16;               // 3125
  const int gruBlks  = (gruTiles + 7) / 8;    // 391
  for (int t = 0; t < TT; ++t) {
    gru_step_k<<<gruBlks, 256, 0, stream>>>(gi + (size_t)t * NN * 192, WhhPk, b_hh,
                                            h, gruTiles);
  }

  // ---- classifier
  classifier_k<<<(NN + 255) / 256, 256, 0, stream>>>(h, Wc1, bc1, Wc2, bc2, out);
}
